// GRAN_73237782331933
// MI455X (gfx1250) — matrix operand order repaired and checked by reading, not compiled, not run on hardware
//
#include <hip/hip_runtime.h>

#define N_NODES 50000
#define N_EDGES 1000000
#define E_TOT   1050000   // edges + self loops
#define N_IMAG  200000
#define N_BLOCK 5000
#define HID     128

typedef __attribute__((ext_vector_type(16))) __bf16 v16bf;
typedef __attribute__((ext_vector_type(8)))  float  v8f;
typedef unsigned short ushort_t;

union ABu { uint4 u[2]; v16bf v; };

__device__ __forceinline__ ushort_t f2bf(float f) {
  union { float f; unsigned int u; } c; c.f = f;
  unsigned int r = c.u + 0x7FFFu + ((c.u >> 16) & 1u);
  return (ushort_t)(r >> 16);
}
__device__ __forceinline__ float bf2f(ushort_t h) {
  union { unsigned int u; float f; } c; c.u = ((unsigned int)h) << 16;
  return c.f;
}

// ---------------------------------------------------------------------------
// Generic bf16 WMMA GEMM:  C[M,N] = act( A[M,K] * Wt[N,K]^T + bias )
// one wave -> one 16x16 output tile, K-loop in steps of 32.
// A-fetch modes: plain (g0==g1==0), row-gather (g0), concat-gather (g0,g1,K=256)
// ---------------------------------------------------------------------------
__global__ __launch_bounds__(256) void gran_wmma_gemm(
    const ushort_t* __restrict__ A, const ushort_t* __restrict__ Wt,
    const float* __restrict__ bias, float* __restrict__ Cf,
    ushort_t* __restrict__ Cb, int M, int N, int K, int relu,
    const int* __restrict__ g0, const int* __restrict__ g1)
{
  const int lane = threadIdx.x & 31;
  const int wave = threadIdx.x >> 5;
  const int half = lane >> 4;          // 0: lanes 0-15, 1: lanes 16-31
  const int l16  = lane & 15;
  const int row0 = blockIdx.x * 16;
  const int n    = wave * 16 + l16;    // output column

  int ra = row0 + l16;
  if (ra >= M) ra = M - 1;             // tail rows: duplicate, stores guarded

  const ushort_t* arow  = nullptr;
  const ushort_t* arow2 = nullptr;
  if (g0 == nullptr) {
    arow = A + (size_t)ra * K;
  } else if (g1 == nullptr) {
    arow = A + (size_t)g0[ra] * K;                 // gather (K == HID)
  } else {
    arow  = A + (size_t)g0[ra] * HID;              // concat halves (K == 256)
    arow2 = A + (size_t)g1[ra] * HID;
  }
  const ushort_t* brow = Wt + (size_t)n * K;

  v8f acc = {};
  for (int kb = 0; kb < K; kb += 32) {
    // A fragment: lane holds row l16, K = kb + half*16 + {0..15}
    const int off = half * 16;
    int ka0 = kb + off, ka1 = kb + off + 8;
    const ushort_t *p0, *p1;
    if (g1 == nullptr) { p0 = arow + ka0; p1 = arow + ka1; }
    else {
      p0 = (ka0 < HID) ? (arow + ka0) : (arow2 + (ka0 - HID));
      p1 = (ka1 < HID) ? (arow + ka1) : (arow2 + (ka1 - HID));
    }
    ABu a, b;
    a.u[0] = *(const uint4*)p0;
    a.u[1] = *(const uint4*)p1;
    // B fragment: lane holds col l16, K = kb + half*16 + {0..15}
    const ushort_t* pb = brow + kb + half * 16;
    b.u[0] = *(const uint4*)pb;
    b.u[1] = *(const uint4*)(pb + 8);
    acc = __builtin_amdgcn_wmma_f32_16x16x32_bf16(
        false, a.v, false, b.v, (short)0, acc, false, false);
  }

  const float bs = bias ? bias[n] : 0.0f;
#pragma unroll
  for (int v = 0; v < 8; ++v) {        // VGPR v -> M = v + half*8
    int m = row0 + v + half * 8;
    if (m < M) {
      float val = acc[v] + bs;
      if (relu) val = fmaxf(val, 0.0f);
      size_t o = (size_t)m * N + n;
      if (Cf) Cf[o] = val;
      if (Cb) Cb[o] = f2bf(val);
    }
  }
}

// ---------------------------------------------------------------------------
// Prep / utility kernels
// ---------------------------------------------------------------------------
__global__ void gran_transpose_bf16(const float* __restrict__ W,
                                    ushort_t* __restrict__ Wt, int K, int N) {
  int i = blockIdx.x * blockDim.x + threadIdx.x;
  if (i >= K * N) return;
  int k = i / N, n = i - k * N;
  Wt[(size_t)n * K + k] = f2bf(W[i]);
}

__global__ void gran_concat_inputs(const float* __restrict__ x,
                                   const float* __restrict__ emb,
                                   ushort_t* __restrict__ A0) {
  int i = blockIdx.x * blockDim.x + threadIdx.x;
  if (i >= N_NODES * HID) return;
  int nd = i >> 7, c = i & 127;
  float v = (c < 64) ? x[nd * 64 + c] : emb[nd * 64 + (c - 64)];
  A0[i] = f2bf(v);
}

__global__ void gran_fill_f32(float* __restrict__ p, float v, int n) {
  int i = blockIdx.x * blockDim.x + threadIdx.x;
  if (i < n) p[i] = v;
}

__global__ void gran_init_gat_out(float* __restrict__ hout,
                                  const float* __restrict__ b) {
  int i = blockIdx.x * blockDim.x + threadIdx.x;
  if (i < N_NODES * HID) hout[i] = b[i & 127];
}

__global__ void gran_f32_to_bf16(const float* __restrict__ s,
                                 ushort_t* __restrict__ d, int n) {
  int i = blockIdx.x * blockDim.x + threadIdx.x;
  if (i < n) d[i] = f2bf(s[i]);
}

// ---------------------------------------------------------------------------
// GAT edge kernels (wave per edge, float4 row traffic, global f32 atomics)
// ---------------------------------------------------------------------------
__device__ __forceinline__ void atomicMaxF(float* addr, float val) {
  if (val >= 0.0f) atomicMax((int*)addr, __float_as_int(val));
  else             atomicMin((unsigned int*)addr, (unsigned int)__float_as_int(val));
}

__global__ __launch_bounds__(256) void gran_edge_score(
    const float* __restrict__ xl, const float* __restrict__ xr,
    const int* __restrict__ ei, const float* __restrict__ att,
    float* __restrict__ e, float* __restrict__ emax)
{
  int edge = blockIdx.x * (blockDim.x >> 5) + (threadIdx.x >> 5);
  if (edge >= E_TOT) return;
  int lane = threadIdx.x & 31;
  int s, d;
  if (edge < N_EDGES) { s = ei[edge]; d = ei[N_EDGES + edge]; }
  else                { s = d = edge - N_EDGES; }           // self loop
  float4 a  = *((const float4*)(xl + (size_t)s * HID) + lane);
  float4 b  = *((const float4*)(xr + (size_t)d * HID) + lane);
  float4 at = *((const float4*)att + lane);
  float m, sum = 0.0f;
  m = a.x + b.x; m = (m > 0.0f) ? m : 0.2f * m; sum += m * at.x;
  m = a.y + b.y; m = (m > 0.0f) ? m : 0.2f * m; sum += m * at.y;
  m = a.z + b.z; m = (m > 0.0f) ? m : 0.2f * m; sum += m * at.z;
  m = a.w + b.w; m = (m > 0.0f) ? m : 0.2f * m; sum += m * at.w;
#pragma unroll
  for (int off = 16; off; off >>= 1) sum += __shfl_xor(sum, off, 32);
  if (lane == 0) { e[edge] = sum; atomicMaxF(emax + d, sum); }
}

__global__ void gran_edge_exp(const float* __restrict__ emax,
                              float* __restrict__ e,
                              const int* __restrict__ ei,
                              float* __restrict__ denom)
{
  int i = blockIdx.x * blockDim.x + threadIdx.x;
  if (i >= E_TOT) return;
  int d = (i < N_EDGES) ? ei[N_EDGES + i] : (i - N_EDGES);
  float ex = __expf(e[i] - emax[d]);
  e[i] = ex;                                   // in-place e -> ex
  atomicAdd(denom + d, ex);
}

__global__ __launch_bounds__(256) void gran_edge_aggregate(
    const float* __restrict__ xl, const float* __restrict__ ex,
    const float* __restrict__ denom, const int* __restrict__ ei,
    float* __restrict__ hout)
{
  int edge = blockIdx.x * (blockDim.x >> 5) + (threadIdx.x >> 5);
  if (edge >= E_TOT) return;
  int lane = threadIdx.x & 31;
  int s, d;
  if (edge < N_EDGES) { s = ei[edge]; d = ei[N_EDGES + edge]; }
  else                { s = d = edge - N_EDGES; }
  float alpha = ex[edge] / denom[d];
  float4 v = *((const float4*)(xl + (size_t)s * HID) + lane);
  float* o = hout + (size_t)d * HID + lane * 4;
  atomicAdd(o + 0, alpha * v.x);
  atomicAdd(o + 1, alpha * v.y);
  atomicAdd(o + 2, alpha * v.z);
  atomicAdd(o + 3, alpha * v.w);
}

// Edge-decoder last layer: N=1 -> wave-per-row dot + sigmoid
__global__ __launch_bounds__(256) void gran_edec_final(
    const ushort_t* __restrict__ t2, const float* __restrict__ W2,
    const float* __restrict__ b2, float* __restrict__ out, int M)
{
  int row = blockIdx.x * (blockDim.x >> 5) + (threadIdx.x >> 5);
  if (row >= M) return;
  int lane = threadIdx.x & 31;
  const ushort_t* p = t2 + (size_t)row * HID + lane * 4;
  float s = 0.0f;
#pragma unroll
  for (int j = 0; j < 4; ++j) s += bf2f(p[j]) * W2[lane * 4 + j];
#pragma unroll
  for (int off = 16; off; off >>= 1) s += __shfl_xor(s, off, 32);
  if (lane == 0) out[row] = 1.0f / (1.0f + __expf(-(s + b2[0])));
}

// ---------------------------------------------------------------------------
extern "C" void kernel_launch(void* const* d_in, const int* in_sizes, int n_in,
                              void* d_out, int out_size, void* d_ws, size_t ws_size,
                              hipStream_t stream) {
  (void)in_sizes; (void)n_in; (void)out_size; (void)ws_size;
  const float* x    = (const float*)d_in[0];
  const float* emb  = (const float*)d_in[1];
  const int*   ei   = (const int*)d_in[2];
  const int*   blki = (const int*)d_in[3];
  const int*   eim  = (const int*)d_in[4];
  const float* encW = (const float*)d_in[5];
  const float* encB = (const float*)d_in[6];
  const float* gWl  = (const float*)d_in[7];
  const float* gBl  = (const float*)d_in[8];
  const float* gWr  = (const float*)d_in[9];
  const float* gBr  = (const float*)d_in[10];
  const float* gAtt = (const float*)d_in[11];
  const float* gB   = (const float*)d_in[12];
  const float* edW0 = (const float*)d_in[13];
  const float* edB0 = (const float*)d_in[14];
  const float* edW1 = (const float*)d_in[15];
  const float* edB1 = (const float*)d_in[16];
  const float* edW2 = (const float*)d_in[17];
  const float* edB2 = (const float*)d_in[18];
  const float* ndW0 = (const float*)d_in[19];
  const float* ndB0 = (const float*)d_in[20];
  const float* ndW1 = (const float*)d_in[21];
  const float* ndB1 = (const float*)d_in[22];
  const float* ndW2 = (const float*)d_in[23];
  const float* ndB2 = (const float*)d_in[24];
  float* out = (float*)d_out;

  // ---- workspace layout ----
  size_t off = 0;
  auto alloc = [&](size_t bytes) -> void* {
    void* p = (char*)d_ws + off; off += (bytes + 255) & ~(size_t)255; return p;
  };
  ushort_t* encWt = (ushort_t*)alloc(3 * 16384 * 2);
  ushort_t* gWlT  = (ushort_t*)alloc(3 * 16384 * 2);
  ushort_t* gWrT  = (ushort_t*)alloc(3 * 16384 * 2);
  ushort_t* edW0t = (ushort_t*)alloc(32768 * 2);       // [128][256]
  ushort_t* edW1t = (ushort_t*)alloc(16384 * 2);
  ushort_t* ndW0t = (ushort_t*)alloc(16384 * 2);
  ushort_t* ndW1t = (ushort_t*)alloc(16384 * 2);
  ushort_t* ndW2t = (ushort_t*)alloc(8192 * 2);        // [64][128]
  ushort_t* abuf0 = (ushort_t*)alloc((size_t)N_NODES * HID * 2);
  ushort_t* abuf1 = (ushort_t*)alloc((size_t)N_NODES * HID * 2);  // hbf lives here
  float* xl    = (float*)alloc((size_t)N_NODES * HID * 4);
  float* xr    = (float*)alloc((size_t)N_NODES * HID * 4);
  float* hout  = (float*)alloc((size_t)N_NODES * HID * 4);
  float* ebuf  = (float*)alloc((size_t)E_TOT * 4);
  float* emax  = (float*)alloc((size_t)N_NODES * 4);
  float* denom = (float*)alloc((size_t)N_NODES * 4);
  ushort_t* tb1 = (ushort_t*)alloc((size_t)N_BLOCK * HID * 2);
  ushort_t* tb2 = (ushort_t*)alloc((size_t)N_BLOCK * HID * 2);
  ushort_t* te1 = (ushort_t*)alloc((size_t)N_IMAG * HID * 2);
  ushort_t* te2 = (ushort_t*)alloc((size_t)N_IMAG * HID * 2);

  auto gemm = [&](const ushort_t* A, const ushort_t* Wt, const float* bias,
                  float* Cf, ushort_t* Cb, int M, int N, int K, int relu,
                  const int* g0, const int* g1) {
    dim3 blk(32 * (N / 16));
    dim3 grd((M + 15) / 16);
    gran_wmma_gemm<<<grd, blk, 0, stream>>>(A, Wt, bias, Cf, Cb, M, N, K, relu, g0, g1);
  };
  auto transp = [&](const float* W, ushort_t* Wt, int K, int N) {
    gran_transpose_bf16<<<(K * N + 255) / 256, 256, 0, stream>>>(W, Wt, K, N);
  };

  // ---- weight prep (fp32 -> bf16, [K][N] -> [N][K]) ----
  for (int i = 0; i < 3; ++i) {
    transp(encW + i * 16384, encWt + i * 16384, HID, HID);
    transp(gWl  + i * 16384, gWlT  + i * 16384, HID, HID);
    transp(gWr  + i * 16384, gWrT  + i * 16384, HID, HID);
  }
  transp(edW0, edW0t, 256, HID);
  transp(edW1, edW1t, HID, HID);
  transp(ndW0, ndW0t, HID, HID);
  transp(ndW1, ndW1t, HID, HID);
  transp(ndW2, ndW2t, HID, 64);

  // ---- encoder MLP: concat(x, emb) -> 128 -> 128 -> 128 ----
  gran_concat_inputs<<<(N_NODES * HID + 255) / 256, 256, 0, stream>>>(x, emb, abuf0);
  gemm(abuf0, encWt,             encB,       nullptr, abuf1, N_NODES, HID, HID, 1, nullptr, nullptr);
  gemm(abuf1, encWt + 16384,     encB + 128, nullptr, abuf0, N_NODES, HID, HID, 1, nullptr, nullptr);
  gemm(abuf0, encWt + 2 * 16384, encB + 256, nullptr, abuf1, N_NODES, HID, HID, 0, nullptr, nullptr);
  ushort_t* hbf = abuf1;

  // ---- 3 GATv2 layers ----
  const int wavesPerBlk = 8;                 // 256 threads
  const int edgeGrid = E_TOT / wavesPerBlk;  // 1050000 / 8 = 131250
  for (int l = 0; l < 3; ++l) {
    gemm(hbf, gWlT + l * 16384, gBl + l * 128, xl, nullptr, N_NODES, HID, HID, 0, nullptr, nullptr);
    gemm(hbf, gWrT + l * 16384, gBr + l * 128, xr, nullptr, N_NODES, HID, HID, 0, nullptr, nullptr);
    gran_fill_f32<<<(N_NODES + 255) / 256, 256, 0, stream>>>(emax, -1.0e30f, N_NODES);
    gran_fill_f32<<<(N_NODES + 255) / 256, 256, 0, stream>>>(denom, 0.0f, N_NODES);
    gran_init_gat_out<<<(N_NODES * HID + 255) / 256, 256, 0, stream>>>(hout, gB + l * 128);
    gran_edge_score<<<edgeGrid, 256, 0, stream>>>(xl, xr, ei, gAtt + l * 128, ebuf, emax);
    gran_edge_exp<<<(E_TOT + 255) / 256, 256, 0, stream>>>(emax, ebuf, ei, denom);
    gran_edge_aggregate<<<edgeGrid, 256, 0, stream>>>(xl, ebuf, denom, ei, hout);
    gran_f32_to_bf16<<<(N_NODES * HID + 255) / 256, 256, 0, stream>>>(hout, hbf, N_NODES * HID);
  }

  // ---- node decoder: h[block] -> 128 -> 128 -> 64 (into d_out[0:320000]) ----
  gemm(hbf, ndW0t, ndB0, nullptr, tb1, N_BLOCK, HID, HID, 1, blki, nullptr);
  gemm(tb1, ndW1t, ndB1, nullptr, tb2, N_BLOCK, HID, HID, 1, nullptr, nullptr);
  gemm(tb2, ndW2t, ndB2, out,     nullptr, N_BLOCK, 64, HID, 0, nullptr, nullptr);

  // ---- edge decoder: concat(h[e0], h[e1]) -> 128 -> 128 -> sigmoid(dot) ----
  gemm(hbf, edW0t, edB0, nullptr, te1, N_IMAG, HID, 256, 1, eim, eim + N_IMAG);
  gemm(te1, edW1t, edB1, nullptr, te2, N_IMAG, HID, HID, 1, nullptr, nullptr);
  gran_edec_final<<<(N_IMAG + 7) / 8, 256, 0, stream>>>(
      te2, edW2, edB2, out + (size_t)N_BLOCK * 64, N_IMAG);
}
